// TransformerBlock_89180700934649
// MI455X (gfx1250) — matrix operand order repaired and checked by reading, not compiled, not run on hardware
//
#include <hip/hip_runtime.h>
#include <hip/hip_bf16.h>

typedef __attribute__((ext_vector_type(16))) __bf16 v16bf;
typedef __attribute__((ext_vector_type(8)))  __bf16 v8bf;
typedef __attribute__((ext_vector_type(8)))  float  v8f;

// ---------------------------------------------------------------------------
// CDNA5 async global->LDS copy (ASYNCcnt path) via inline asm.
// (The clang builtin exists but its pointer params are in the unspellable
//  cuda_device/cuda_shared language address spaces -> use the ISA directly.)
// VFLAT async encoding: VDST = LDS byte address VGPR, VADDR = 64-bit global
// address, SADDR = off. Completion tracked by ASYNCcnt.
// ---------------------------------------------------------------------------
__device__ __forceinline__ void async_copy16(const __bf16* __restrict__ g,
                                             __bf16* l) {
#if defined(__AMDGCN__)
    const unsigned lds_addr = (unsigned)(unsigned long long)l;  // addr[31:0] = LDS offset
    const unsigned long long gaddr = (unsigned long long)g;
    asm volatile("global_load_async_to_lds_b128 %0, %1, off"
                 :
                 : "v"(lds_addr), "v"(gaddr)
                 : "memory");
#else
    *(v8bf*)l = *(const v8bf*)g;
#endif
}

template <int N>
__device__ __forceinline__ void wait_async_le() {
#if defined(__AMDGCN__)
#if __has_builtin(__builtin_amdgcn_s_wait_asynccnt)
    __builtin_amdgcn_s_wait_asynccnt(N);
#else
    asm volatile("s_wait_asynccnt %0" ::"n"(N) : "memory");
#endif
#endif
}

// ---------------------------------------------------------------------------
// Split-precision helpers: x ~= hi(bf16) + lo(bf16 of residual).
// GEMM math: acc += ah*bh + ah*bl + al*bh  (f32 accumulate, 3x WMMA)
// ---------------------------------------------------------------------------
__device__ __forceinline__ void bf16_split(float x, __bf16& h, __bf16& l) {
    h = (__bf16)x;
    l = (__bf16)(x - (float)h);
}

__device__ __forceinline__ v8f wmma3(v16bf ah, v16bf al, v16bf bh, v16bf bl, v8f c) {
    c = __builtin_amdgcn_wmma_f32_16x16x32_bf16(false, ah, false, bh, (short)0, c, false, false);
    c = __builtin_amdgcn_wmma_f32_16x16x32_bf16(false, ah, false, bl, (short)0, c, false, false);
    c = __builtin_amdgcn_wmma_f32_16x16x32_bf16(false, al, false, bh, (short)0, c, false, false);
    return c;
}

// ---------------------------------------------------------------------------
// Fragment loaders from pre-split bf16 planes (wave32; A and B use the same contiguous k order).
// A operand 16x32 (MxK): lane L: row m=L%16, half=L/16;
//   vector elements 0..7  = k0 + half*16 + (0..7)
//   vector elements 8..15 = k0 + half*16 + 8 + (0..7)      -> two 16B loads
// B operand 32x16 (KxN): lane L: col n=L%16, half=L/16;
//   vector elements 0..15 = k0 + half*16 + (0..15)          -> one 32B load
// C/D: vgpr r, lane L: row = r + (L/16)*8, col = L%16
// ---------------------------------------------------------------------------
__device__ __forceinline__ v16bf load_a_plane(const __bf16* __restrict__ src,
                                              int m0, int k0, int ld) {
    const int lane = threadIdx.x & 31;
    const int m = lane & 15, half = lane >> 4;
    const __bf16* p = src + (size_t)(m0 + m) * ld + k0 + half * 16;
    const v8bf a = *(const v8bf*)p;
    const v8bf b = *(const v8bf*)(p + 8);
    return __builtin_shufflevector(a, b, 0, 1, 2, 3, 4, 5, 6, 7,
                                   8, 9, 10, 11, 12, 13, 14, 15);
}

__device__ __forceinline__ v16bf load_b_plane(const __bf16* __restrict__ src,
                                              int n0, int k0, int ld) {
    const int lane = threadIdx.x & 31;
    const int n = lane & 15, half = lane >> 4;
    const __bf16* p = src + (size_t)(n0 + n) * ld + k0 + half * 16;
    return *(const v16bf*)p;
}

// ---------------------------------------------------------------------------
// One-shot f32 -> (hi, lo) bf16 plane conversion (weights)
// ---------------------------------------------------------------------------
__global__ __launch_bounds__(256)
void split_kernel(const float* __restrict__ src, __bf16* __restrict__ hi,
                  __bf16* __restrict__ lo, int n) {
    const int i = blockIdx.x * 256 + threadIdx.x;
    if (i < n) {
        __bf16 h, l;
        bf16_split(src[i], h, l);
        hi[i] = h;
        lo[i] = l;
    }
}

// ---------------------------------------------------------------------------
// LayerNorm: one 256-thread WG per row; writes bf16 hi/lo planes
// ---------------------------------------------------------------------------
__global__ __launch_bounds__(256)
void ln_split_kernel(const float* __restrict__ x, const float* __restrict__ g,
                     const float* __restrict__ b, __bf16* __restrict__ yH,
                     __bf16* __restrict__ yL, int C) {
    __shared__ float red[16];
    __shared__ float smu, srstd;
    const size_t row = blockIdx.x;
    const float* xr = x + row * C;
    float s = 0.f, s2 = 0.f;
    for (int c = threadIdx.x; c < C; c += 256) {
        const float v = xr[c];
        s += v; s2 += v * v;
    }
    const int lane = threadIdx.x & 31, wave = threadIdx.x >> 5;
#pragma unroll
    for (int off = 16; off > 0; off >>= 1) {
        s  += __shfl_down(s,  off, 32);
        s2 += __shfl_down(s2, off, 32);
    }
    if (lane == 0) { red[wave] = s; red[8 + wave] = s2; }
    __syncthreads();
    if (threadIdx.x == 0) {
        float ts = 0.f, ts2 = 0.f;
#pragma unroll
        for (int i = 0; i < 8; ++i) { ts += red[i]; ts2 += red[8 + i]; }
        const float mu  = ts / (float)C;
        const float var = ts2 / (float)C - mu * mu;
        smu = mu;
        srstd = rsqrtf(var + 1e-5f);
    }
    __syncthreads();
    const float mu = smu, rs = srstd;
    for (int c = threadIdx.x; c < C; c += 256) {
        const float y = (xr[c] - mu) * rs * g[c] + b[c];
        __bf16 h, l;
        bf16_split(y, h, l);
        yH[row * C + c] = h;
        yL[row * C + c] = l;
    }
}

// ---------------------------------------------------------------------------
// GEMM: Y(MxN) = act( A(MxK) * W(NxK)^T + bias ), operands = bf16 hi/lo
// planes, f32 accumulate. 128x128 tile / WG (8 waves, wave = 32x64 subtile).
// K-slices (128x32) of A and W are double-buffer staged into LDS with
// async global->LDS copies (ASYNCcnt) and consumed by WMMA from LDS.
// Output either f32 (+resid) or bf16 hi/lo planes (optionally transposed to
// (b, N, T) layout for the attention V operand).
// ---------------------------------------------------------------------------
__global__ __launch_bounds__(256)
void gemm_bf16_kernel(const __bf16* __restrict__ AH, const __bf16* __restrict__ AL,
                      const __bf16* __restrict__ WH, const __bf16* __restrict__ WL,
                      const float* __restrict__ bias, const float* __restrict__ resid,
                      float* __restrict__ Yf, __bf16* __restrict__ YH,
                      __bf16* __restrict__ YL, int Ndim, int Kdim,
                      int relu_flag, int transT) {
    __shared__ __align__(64) __bf16 sAH[2][4096];   // 128 rows x 32 k
    __shared__ __align__(64) __bf16 sAL[2][4096];
    __shared__ __align__(64) __bf16 sBH[2][4096];
    __shared__ __align__(64) __bf16 sBL[2][4096];
    const int tid  = threadIdx.x;
    const int lane = tid & 31;
    const int wave = tid >> 5;
    const int wrow = wave >> 1, wcol = wave & 1;
    const int rowB = blockIdx.y * 128;
    const int colB = blockIdx.x * 128;

    // stage K-slice [k0, k0+32) of the 128-row A and B tiles into buffer `buf`
    // (512 16-byte chunks per plane; 2 per thread; 8 async ops per wave)
    auto stage = [&](int buf, int k0) {
#pragma unroll
        for (int i = 0; i < 2; ++i) {
            const int c  = i * 256 + tid;
            const int r  = c >> 2;            // tile row 0..127
            const int ko = (c & 3) * 8;       // k offset 0,8,16,24
            const int lofs = r * 32 + ko;
            const size_t ga = (size_t)(rowB + r) * Kdim + k0 + ko;
            const size_t gb = (size_t)(colB + r) * Kdim + k0 + ko;
            async_copy16(AH + ga, &sAH[buf][lofs]);
            async_copy16(AL + ga, &sAL[buf][lofs]);
            async_copy16(WH + gb, &sBH[buf][lofs]);
            async_copy16(WL + gb, &sBL[buf][lofs]);
        }
    };

    v8f acc[2][4] = {};
    stage(0, 0);
    for (int k0 = 0; k0 < Kdim; k0 += 32) {
        const int cur = (k0 >> 5) & 1;
        const bool more = (k0 + 32) < Kdim;
        if (more) {
            stage(cur ^ 1, k0 + 32);   // prefetch next slice into other buffer
            wait_async_le<8>();        // retire the 8 ops that filled `cur`
        } else {
            wait_async_le<0>();
        }
        __syncthreads();               // all waves' copies into `cur` visible
        v16bf ah[2], al[2], bh[4], bl[4];
#pragma unroll
        for (int i = 0; i < 2; ++i) {
            ah[i] = load_a_plane(sAH[cur], wrow * 32 + i * 16, 0, 32);
            al[i] = load_a_plane(sAL[cur], wrow * 32 + i * 16, 0, 32);
        }
#pragma unroll
        for (int j = 0; j < 4; ++j) {
            bh[j] = load_b_plane(sBH[cur], wcol * 64 + j * 16, 0, 32);
            bl[j] = load_b_plane(sBL[cur], wcol * 64 + j * 16, 0, 32);
        }
#pragma unroll
        for (int i = 0; i < 2; ++i)
#pragma unroll
            for (int j = 0; j < 4; ++j)
                acc[i][j] = wmma3(ah[i], al[i], bh[j], bl[j], acc[i][j]);
        __syncthreads();               // protect `cur` before it is re-staged
    }

    const int half = lane >> 4, nn = lane & 15;
#pragma unroll
    for (int i = 0; i < 2; ++i)
#pragma unroll
        for (int j = 0; j < 4; ++j)
#pragma unroll
            for (int r = 0; r < 8; ++r) {
                const int row = rowB + wrow * 32 + i * 16 + r + half * 8;
                const int col = colB + wcol * 64 + j * 16 + nn;
                float v = acc[i][j][r];
                if (bias) v += bias[col];
                if (relu_flag) v = fmaxf(v, 0.f);
                if (Yf) {
                    const size_t idx = (size_t)row * Ndim + col;
                    Yf[idx] = resid ? v + resid[idx] : v;
                } else {
                    size_t idx;
                    if (transT) {
                        const int bb = row / transT;
                        const int t  = row - bb * transT;
                        idx = (size_t)bb * Ndim * transT + (size_t)col * transT + t;
                    } else {
                        idx = (size_t)row * Ndim + col;
                    }
                    __bf16 h, l;
                    bf16_split(v, h, l);
                    YH[idx] = h;
                    YL[idx] = l;
                }
            }
}

// ---------------------------------------------------------------------------
// Flash attention (causal). One WG (8 waves) per (batch, 16 query rows).
// QK^T partials per wave (K split 8 ways) reduced via ds_add_f32 into LDS;
// wave0 does online softmax; P stored in LDS as bf16 hi/lo; P@V via WMMA
// with V pre-transposed to (b, C, T). Fused residual: xout = xin + attn.
// ---------------------------------------------------------------------------
__global__ __launch_bounds__(256)
void attn_kernel(const __bf16* __restrict__ qH, const __bf16* __restrict__ qL,
                 const __bf16* __restrict__ kH, const __bf16* __restrict__ kL,
                 const __bf16* __restrict__ vTH, const __bf16* __restrict__ vTL,
                 const float* __restrict__ xin, float* __restrict__ xout,
                 int T, int C) {
    const int bidx = blockIdx.y;
    const int t0 = blockIdx.x * 16;
    const int lane = threadIdx.x & 31, wave = threadIdx.x >> 5;
    const int half = lane >> 4, l15 = lane & 15;
    __shared__ float S[512];                       // 16 x 32 scores (f32)
    __shared__ __align__(64) __bf16 PH[512];       // 16 x 32 probs hi
    __shared__ __align__(64) __bf16 PL[512];       // 16 x 32 probs lo
    __shared__ float mrow[16], lrow[16], crow[16];
    const size_t bo = (size_t)bidx * T * C;        // same extent for q/k and vT
    const __bf16* qHb = qH + bo;  const __bf16* qLb = qL + bo;
    const __bf16* kHb = kH + bo;  const __bf16* kLb = kL + bo;
    const __bf16* vHb = vTH + bo; const __bf16* vLb = vTL + bo;
    v8f acc[8] = {};
    if (threadIdx.x < 16) { mrow[threadIdx.x] = -1e30f; lrow[threadIdx.x] = 0.f; }
    const float scale = 0.03125f;                  // 1/sqrt(1024)
    const int nkb = (t0 + 47) >> 5;                // 32-key blocks covering [0, t0+15]
    for (int kblk = 0; kblk < nkb; ++kblk) {
        const int kbase = kblk * 32;
        for (int i = threadIdx.x; i < 512; i += 256) S[i] = 0.f;
        __syncthreads();
        // --- partial scores: wave w covers feature slice [w*128, w*128+128) ---
        v8f s0 = {}, s1 = {};
#pragma unroll
        for (int kk = 0; kk < 4; ++kk) {
            const int c0 = wave * 128 + kk * 32;
            const v16bf qh = load_a_plane(qHb, t0, c0, C);
            const v16bf ql = load_a_plane(qLb, t0, c0, C);
            v16bf kh = load_b_plane(kHb, kbase, c0, C);
            v16bf kl = load_b_plane(kLb, kbase, c0, C);
            s0 = wmma3(qh, ql, kh, kl, s0);
            kh = load_b_plane(kHb, kbase + 16, c0, C);
            kl = load_b_plane(kLb, kbase + 16, c0, C);
            s1 = wmma3(qh, ql, kh, kl, s1);
        }
#pragma unroll
        for (int r = 0; r < 8; ++r) {
            atomicAdd(&S[(r + half * 8) * 32 + l15], s0[r]);
            atomicAdd(&S[(r + half * 8) * 32 + 16 + l15], s1[r]);
        }
        __syncthreads();
        // --- online softmax update (wave0 lanes 0..15, one row each) ---
        if (threadIdx.x < 16) {
            const int i = threadIdx.x;
            const int t = t0 + i;
            const float mold = mrow[i];
            float mnew = mold;
#pragma unroll
            for (int j = 0; j < 32; ++j) {
                const float sv = (kbase + j <= t) ? S[i * 32 + j] * scale : -1e30f;
                mnew = fmaxf(mnew, sv);
            }
            const float corr = __expf(mold - mnew);
            float lsum = lrow[i] * corr;
#pragma unroll
            for (int j = 0; j < 32; ++j) {
                const float sv = (kbase + j <= t) ? S[i * 32 + j] * scale : -1e30f;
                const float p = __expf(sv - mnew);
                __bf16 h, l;
                bf16_split(p, h, l);
                PH[i * 32 + j] = h;
                PL[i * 32 + j] = l;
                lsum += p;
            }
            mrow[i] = mnew; lrow[i] = lsum; crow[i] = corr;
        }
        __syncthreads();
        // --- rescale accumulators, then acc += P @ V ---
#pragma unroll
        for (int r = 0; r < 8; ++r) {
            const float cf = crow[r + half * 8];
#pragma unroll
            for (int nt = 0; nt < 8; ++nt) acc[nt][r] *= cf;
        }
        const v16bf ph = load_a_plane(PH, 0, 0, 32);
        const v16bf pl = load_a_plane(PL, 0, 0, 32);
#pragma unroll
        for (int nt = 0; nt < 8; ++nt) {
            const int col = wave * 128 + nt * 16;
            const v16bf vh = load_b_plane(vHb, col, kbase, T);
            const v16bf vl = load_b_plane(vLb, col, kbase, T);
            acc[nt] = wmma3(ph, pl, vh, vl, acc[nt]);
        }
        __syncthreads();
    }
    // --- epilogue: normalize by l, add residual ---
    const float* xb = xin + bo;
    float* ob = xout + bo;
#pragma unroll
    for (int r = 0; r < 8; ++r) {
        const int m = r + half * 8;
        const float inv_l = 1.f / lrow[m];
#pragma unroll
        for (int nt = 0; nt < 8; ++nt) {
            const int col = wave * 128 + nt * 16 + l15;
            const size_t idx = (size_t)(t0 + m) * C + col;
            ob[idx] = xb[idx] + acc[nt][r] * inv_l;
        }
    }
}

// ---------------------------------------------------------------------------
// Host side
// ---------------------------------------------------------------------------
extern "C" void kernel_launch(void* const* d_in, const int* in_sizes, int n_in,
                              void* d_out, int out_size, void* d_ws, size_t ws_size,
                              hipStream_t stream) {
    (void)in_sizes; (void)n_in; (void)out_size; (void)ws_size;
    const float* x    = (const float*)d_in[0];
    const float* Wq   = (const float*)d_in[1];
    const float* Wk   = (const float*)d_in[2];
    const float* Wv   = (const float*)d_in[3];
    const float* W1   = (const float*)d_in[4];
    const float* b1   = (const float*)d_in[5];
    const float* W2   = (const float*)d_in[6];
    const float* b2   = (const float*)d_in[7];
    const float* ln1w = (const float*)d_in[8];
    const float* ln1b = (const float*)d_in[9];
    const float* ln2w = (const float*)d_in[10];
    const float* ln2b = (const float*)d_in[11];

    const int B = 4, T = 2048, C = 1024, F = 4096;
    const int MTOT = B * T;                 // 8192
    const size_t MC = (size_t)MTOT * C;     // 8.39M
    const size_t CC = (size_t)C * C;        // 1.05M
    const size_t FC = (size_t)F * C;        // 4.19M
    const size_t MF = (size_t)MTOT * F;     // 33.6M

    // ---- workspace layout (bytes) ----
    char* base = (char*)d_ws;
    size_t off = 0;
    float*  x2  = (float*)(base + off);  off += MC * 4;        // f32 residual
    __bf16* hH  = (__bf16*)(base + off); off += MC * 2;        // LN out hi
    __bf16* hL  = (__bf16*)(base + off); off += MC * 2;        // LN out lo
    __bf16* WqH = (__bf16*)(base + off); off += CC * 2;
    __bf16* WqL = (__bf16*)(base + off); off += CC * 2;
    __bf16* WkH = (__bf16*)(base + off); off += CC * 2;
    __bf16* WkL = (__bf16*)(base + off); off += CC * 2;
    __bf16* WvH = (__bf16*)(base + off); off += CC * 2;
    __bf16* WvL = (__bf16*)(base + off); off += CC * 2;
    __bf16* W1H = (__bf16*)(base + off); off += FC * 2;
    __bf16* W1L = (__bf16*)(base + off); off += FC * 2;
    __bf16* W2H = (__bf16*)(base + off); off += FC * 2;
    __bf16* W2L = (__bf16*)(base + off); off += FC * 2;
    // activation region: q/k/vT planes live here during attention; the ff
    // planes reuse the same region afterwards (attention has consumed q/k/v).
    const size_t act = off;
    __bf16* qH  = (__bf16*)(base + act + 0 * MC * 2);
    __bf16* qL  = (__bf16*)(base + act + 1 * MC * 2);
    __bf16* kHp = (__bf16*)(base + act + 2 * MC * 2);
    __bf16* kLp = (__bf16*)(base + act + 3 * MC * 2);
    __bf16* vTH = (__bf16*)(base + act + 4 * MC * 2);
    __bf16* vTL = (__bf16*)(base + act + 5 * MC * 2);
    __bf16* ffH = (__bf16*)(base + act + 0 * MF * 2);
    __bf16* ffL = (__bf16*)(base + act + 1 * MF * 2);

    // ---- 0. split weights into bf16 hi/lo planes (one-shot, bandwidth-trivial)
    const int nb_cc = (int)((CC + 255) / 256), nb_fc = (int)((FC + 255) / 256);
    split_kernel<<<nb_cc, 256, 0, stream>>>(Wq, WqH, WqL, (int)CC);
    split_kernel<<<nb_cc, 256, 0, stream>>>(Wk, WkH, WkL, (int)CC);
    split_kernel<<<nb_cc, 256, 0, stream>>>(Wv, WvH, WvL, (int)CC);
    split_kernel<<<nb_fc, 256, 0, stream>>>(W1, W1H, W1L, (int)FC);
    split_kernel<<<nb_fc, 256, 0, stream>>>(W2, W2H, W2L, (int)FC);

    // ---- 1. h = LN1(x)
    ln_split_kernel<<<MTOT, 256, 0, stream>>>(x, ln1w, ln1b, hH, hL, C);

    // ---- 2. q,k = h @ W^T ; vT = (h @ Wv^T)^T
    dim3 gqkv(C / 128, MTOT / 128);
    gemm_bf16_kernel<<<gqkv, 256, 0, stream>>>(hH, hL, WqH, WqL, nullptr, nullptr,
                                               nullptr, qH, qL, C, C, 0, 0);
    gemm_bf16_kernel<<<gqkv, 256, 0, stream>>>(hH, hL, WkH, WkL, nullptr, nullptr,
                                               nullptr, kHp, kLp, C, C, 0, 0);
    gemm_bf16_kernel<<<gqkv, 256, 0, stream>>>(hH, hL, WvH, WvL, nullptr, nullptr,
                                               nullptr, vTH, vTL, C, C, 0, T);

    // ---- 3. x2 = x + causal_attention(q, k, v)
    attn_kernel<<<dim3(T / 16, B), 256, 0, stream>>>(qH, qL, kHp, kLp, vTH, vTL,
                                                     x, x2, T, C);

    // ---- 4. h = LN2(x2)
    ln_split_kernel<<<MTOT, 256, 0, stream>>>(x2, ln2w, ln2b, hH, hL, C);

    // ---- 5. ff = relu(h @ W1^T + b1)
    gemm_bf16_kernel<<<dim3(F / 128, MTOT / 128), 256, 0, stream>>>(
        hH, hL, W1H, W1L, b1, nullptr, nullptr, ffH, ffL, F, C, 1, 0);

    // ---- 6. out = x2 + ff @ W2^T + b2
    gemm_bf16_kernel<<<dim3(C / 128, MTOT / 128), 256, 0, stream>>>(
        ffH, ffL, W2H, W2L, b2, x2, (float*)d_out, nullptr, nullptr, C, F, 0, 0);
}
